// Model_81192061763615
// MI455X (gfx1250) — compile-verified, matrix operand order checked by reading, not run on hardware
//
#include <hip/hip_runtime.h>

typedef _Float16 h8  __attribute__((ext_vector_type(8)));
typedef _Float16 h16 __attribute__((ext_vector_type(16)));
typedef float    f8  __attribute__((ext_vector_type(8)));
typedef int      v4i __attribute__((ext_vector_type(4)));

#define SEQ   640                 // 64*10 sequences
#define TSTEP 32
#define FF    1024
#define NROW  (SEQ*TSTEP)         // 20480 rows in all GEMMs
#define BN_INV 0.9999950000374997f

#if defined(__has_builtin)
# if __has_builtin(__builtin_amdgcn_global_load_async_to_lds_b128) && \
     __has_builtin(__builtin_amdgcn_s_wait_asynccnt)
#  define USE_ASYNC_LDS 1
# endif
#endif

__device__ __forceinline__ float silu_f(float x){ return x / (1.0f + __expf(-x)); }

__device__ __forceinline__ h16 cat8(h8 lo, h8 hi){
  h16 r;
#pragma unroll
  for (int e = 0; e < 8; ++e){ r[e] = lo[e]; r[e+8] = hi[e]; }
  return r;
}

// ---------------------------------------------------------------------------
// C(MxN) = [act(A(MxK) @ Bt(NxK)^T + bias) * scale] + resid
// A,Bt row-major f16; C written as f32 and/or f16 with row stride ldc.
// M % 128 == 0, K % 32 == 0; N arbitrary (row-clamped loads + guarded stores).
// Block tile 128x128, 8 wave32 waves (4x2); wave tile 32x64 = 2x4 WMMA tiles.
// LDS double-buffered (one barrier per 32-K step); tiles staged either via
// gfx1250 GLOBAL_LOAD_ASYNC_TO_LDS_B128 (ASYNCcnt) or via register staging.
// ---------------------------------------------------------------------------
__global__ __launch_bounds__(256) void wmma_gemm(
    const _Float16* __restrict__ A, const _Float16* __restrict__ Bt,
    const float* __restrict__ bias, const float* __restrict__ resid,
    float* __restrict__ C32, _Float16* __restrict__ C16,
    int M, int N, int K, int ldc, int act, float scale)
{
  __shared__ _Float16 As[2][128*40];   // 128 rows x 32 halves, stride 40 (pad)
  __shared__ _Float16 Bs[2][128*40];
  const int tid  = threadIdx.x;
  const int wid  = tid >> 5, lane = tid & 31;
  const int bm   = blockIdx.y * 128, bn = blockIdx.x * 128;
  const int wr   = wid & 3;            // wave row (0..3)  -> 32 rows each
  const int wc   = wid >> 2;           // wave col (0..1)  -> 64 cols each
  const int frow = lane & 15, fhalf = lane >> 4;
  const int sr0  = tid >> 2;           // staging row 0..63 (+64 for j=1)
  const int sc0  = tid & 3;            // staging 8-half chunk 0..3

  f8 acc[2][4] = {};

  auto stage = [&](int k0, int s){
#pragma unroll
    for (int j = 0; j < 2; ++j){
      int r = sr0 + 64*j;
      int gn = bn + r; if (gn >= N) gn = N - 1;   // clamp: cols >= N discarded
      const _Float16* ga = A  + (size_t)(bm + r)*K + k0 + sc0*8;
      const _Float16* gb = Bt + (size_t)gn*K      + k0 + sc0*8;
      _Float16* la = As[s] + r*40 + sc0*8;
      _Float16* lb = Bs[s] + r*40 + sc0*8;
#ifdef USE_ASYNC_LDS
      __builtin_amdgcn_global_load_async_to_lds_b128(
          (v4i*)ga, (__attribute__((address_space(3))) v4i*)la, 0, 0);
      __builtin_amdgcn_global_load_async_to_lds_b128(
          (v4i*)gb, (__attribute__((address_space(3))) v4i*)lb, 0, 0);
#else
      h8 va = *(const h8*)ga;
      h8 vb = *(const h8*)gb;
      *(h8*)la = va;
      *(h8*)lb = vb;
#endif
    }
  };

  const int nsteps = K >> 5;
  stage(0, 0);
#ifdef USE_ASYNC_LDS
  __builtin_amdgcn_s_wait_asynccnt(0);
#endif
  __syncthreads();

  int s = 0;
  for (int i = 0; i < nsteps; ++i){
    if (i + 1 < nsteps) stage((i + 1) << 5, s ^ 1);   // fill other buffer (DMA overlaps compute)

    h16 af[2], bf[4];
#pragma unroll
    for (int mi = 0; mi < 2; ++mi) {
      const _Float16* p = As[s] + (wr*32 + mi*16 + frow)*40 + fhalf*8;
      af[mi] = cat8(*(const h8*)p, *(const h8*)(p + 16));
    }
#pragma unroll
    for (int nj = 0; nj < 4; ++nj) {
      const _Float16* p = Bs[s] + (wc*64 + nj*16 + frow)*40 + fhalf*8;
      bf[nj] = cat8(*(const h8*)p, *(const h8*)(p + 16));
    }
#pragma unroll
    for (int mi = 0; mi < 2; ++mi)
#pragma unroll
      for (int nj = 0; nj < 4; ++nj)
        acc[mi][nj] = __builtin_amdgcn_wmma_f32_16x16x32_f16(
            false, af[mi], false, bf[nj], (short)0, acc[mi][nj], false, false);

    if (i + 1 < nsteps){
#ifdef USE_ASYNC_LDS
      __builtin_amdgcn_s_wait_asynccnt(0);
#endif
      __syncthreads();
      s ^= 1;
    }
  }

  // Epilogue: C layout — lane holds col = lane%16, VGPR v holds row v + 8*(lane/16)
#pragma unroll
  for (int mi = 0; mi < 2; ++mi) {
#pragma unroll
    for (int nj = 0; nj < 4; ++nj) {
      int gc = bn + wc*64 + nj*16 + frow;
      if (gc >= N) continue;
      float bv = bias ? bias[gc] : 0.0f;
      int gr0 = bm + wr*32 + mi*16 + fhalf*8;
#pragma unroll
      for (int v = 0; v < 8; ++v) {
        int gr = gr0 + v;
        float val = acc[mi][nj][v] + bv;
        if (act == 1) val = fmaxf(val, 0.0f);
        val *= scale;
        if (resid) val += resid[(size_t)gr*ldc + gc];
        if (C32) C32[(size_t)gr*ldc + gc] = val;
        if (C16) C16[(size_t)gr*ldc + gc] = (_Float16)val;
      }
    }
  }
}

// --------------------------- helper kernels --------------------------------
__global__ __launch_bounds__(256) void cvt16(const float* __restrict__ s,
                                             _Float16* __restrict__ d, long n){
  long i = (long)blockIdx.x*256 + threadIdx.x;
  if (i < n) d[i] = (_Float16)s[i];
}

// w[o][f][k] -> dst[o][k*F + f]  (match im2col column order)
__global__ __launch_bounds__(256) void cvt_convw(const float* __restrict__ w,
                                                 _Float16* __restrict__ dst,
                                                 int O, int Fdim, int Kk){
  long idx = (long)blockIdx.x*256 + threadIdx.x;
  long total = (long)O*Fdim*Kk;
  if (idx >= total) return;
  int k = (int)(idx % Kk);
  long r = idx / Kk;
  int f = (int)(r % Fdim);
  int o = (int)(r / Fdim);
  dst[(size_t)o*Fdim*Kk + (size_t)k*Fdim + f] = (_Float16)w[idx];
}

// depthwise causal conv (K=4) over time + SiLU; xz row = xh||gate (2048 wide)
__global__ __launch_bounds__(256) void mamba_mid(const float* __restrict__ xz,
    const float* __restrict__ cw, const float* __restrict__ cb,
    float* __restrict__ xa32, _Float16* __restrict__ xa16){
  long idx = (long)blockIdx.x*256 + threadIdx.x;   // over NROW*FF
  int h = (int)(idx & (FF-1));
  long row = idx >> 10;
  int t = (int)(row & (TSTEP-1));
  float acc = cb[h];
#pragma unroll
  for (int k = 0; k < 4; ++k){
    int tt = t - 3 + k;
    if (tt >= 0) acc += xz[(row - t + tt)*2048 + h] * cw[h*4 + k];
  }
  float s = silu_f(acc);
  xa32[idx] = s;
  xa16[idx] = (_Float16)s;
}

// per-sequence SSM scan: h_t = h_{t-1} @ A^T + u_t, cs_t = h_t . C   (wave32)
__global__ __launch_bounds__(256) void scan_kernel(const float* __restrict__ xs,
    const float* __restrict__ Am, const float* __restrict__ Bv,
    const float* __restrict__ Cv, float* __restrict__ cs){
  int lane = threadIdx.x & 31;
  int b = blockIdx.x*8 + (threadIdx.x >> 5);
  float Arow[32];
#pragma unroll
  for (int j = 0; j < 32; ++j) Arow[j] = Am[lane*32 + j];
  float bb = Bv[lane], cc = Cv[lane];
  float h = 0.0f;
  const float* xsb = xs + (size_t)b*TSTEP*32;
  for (int t = 0; t < TSTEP; ++t){
    float u = xsb[t*32 + lane] * bb;
    float nh = u;
#pragma unroll
    for (int j = 0; j < 32; ++j) nh += Arow[j] * __shfl(h, j, 32);
    h = nh;
    float c = h * cc;
#pragma unroll
    for (int o = 16; o > 0; o >>= 1) c += __shfl_xor(c, o, 32);
    if (lane == 0) cs[b*TSTEP + t] = c;
  }
}

// yg = ((cs + D) * xa) * silu(gate)
__global__ __launch_bounds__(256) void ygate_kernel(const float* __restrict__ cs,
    const float* __restrict__ xa32, const float* __restrict__ Dv,
    const float* __restrict__ xz, _Float16* __restrict__ yg){
  long idx = (long)blockIdx.x*256 + threadIdx.x;
  int h = (int)(idx & (FF-1));
  long row = idx >> 10;
  float xa = xa32[idx];
  float y = cs[row]*xa + Dv[h]*xa;
  float g = xz[row*2048 + 1024 + h];
  yg[idx] = (_Float16)(y * silu_f(g));
}

// im2col for time-conv K=3, pad 1: dst row = [x[t-1], x[t], x[t+1]] (3072)
__global__ __launch_bounds__(256) void im2col3(const _Float16* __restrict__ src,
                                               _Float16* __restrict__ dst){
  long idx = (long)blockIdx.x*256 + threadIdx.x;   // NROW*384 8-half chunks
  long row = idx / 384;
  int c = (int)(idx - row*384);
  int k = c >> 7, f8c = c & 127;
  int t = (int)(row & (TSTEP-1));
  int tt = t + k - 1;
  h8 v = {};
  if (tt >= 0 && tt < TSTEP) v = *(const h8*)(src + (row - t + tt)*FF + f8c*8);
  *(h8*)(dst + row*3072 + (size_t)c*8) = v;
}

__global__ __launch_bounds__(256) void rownorm_kernel(const float* __restrict__ x,
                                                      float* __restrict__ mag){
  int lane = threadIdx.x & 31;
  long row = (long)blockIdx.x*8 + (threadIdx.x >> 5);
  const float* p = x + row*FF;
  float a = 0.0f;
  for (int f = lane; f < FF; f += 32){ float v = p[f]; a += v*v; }
#pragma unroll
  for (int o = 16; o > 0; o >>= 1) a += __shfl_xor(a, o, 32);
  if (lane == 0) mag[row] = sqrtf(a);
}

__global__ __launch_bounds__(256) void fc3_kernel(const _Float16* __restrict__ s2,
    const float* __restrict__ w, const float* __restrict__ b,
    float* __restrict__ s3){
  int lane = threadIdx.x & 31;
  long row = (long)blockIdx.x*8 + (threadIdx.x >> 5);
  const _Float16* p = s2 + row*128;
  float a = 0.0f;
#pragma unroll
  for (int i = 0; i < 4; ++i){ int f = lane + i*32; a += (float)p[f] * w[f]; }
#pragma unroll
  for (int o = 16; o > 0; o >>= 1) a += __shfl_xor(a, o, 32);
  if (lane == 0) s3[row] = 1.0f/(1.0f + __expf(-(a + b[0])));
}

__global__ __launch_bounds__(256) void scores_featmag(const float* __restrict__ s3,
    const float* __restrict__ mag, float* __restrict__ sc, float* __restrict__ fm){
  int idx = blockIdx.x*256 + threadIdx.x;          // 64*32
  if (idx >= 2048) return;
  int bb = idx >> 5, t = idx & 31;
  float a = 0.0f, m = 0.0f;
  for (int c = 0; c < 10; ++c){
    long r = (long)(bb*10 + c)*32 + t;
    a += s3[r]; m += mag[r];
  }
  sc[idx] = a * 0.1f;
  fm[idx] = m * 0.1f;
}

__global__ __launch_bounds__(256) void anomaly_kernel(const float* __restrict__ feat,
    const float* __restrict__ pmem, const float* __restrict__ smem,
    float* __restrict__ an){
  __shared__ float fin[FF];
  __shared__ float ps[256], ss[256], red[256];
  int tid = threadIdx.x, b = blockIdx.x;
  for (int f = tid; f < FF; f += 256)
    fin[f] = feat[((size_t)b*TSTEP + (TSTEP-1))*FF + f];
  __syncthreads();
  float ap = 0.0f, as = 0.0f;
  const float* pp = pmem + (size_t)tid*FF;
  const float* sp = smem + (size_t)tid*FF;
  for (int f = 0; f < FF; ++f){
    float d = pp[f] - fin[f]; ap += d*d;
    float e = sp[f] - fin[f]; as += e*e;
  }
  ps[tid] = sqrtf(ap); ss[tid] = sqrtf(as);
  red[tid] = ps[tid]; __syncthreads();
  for (int o = 128; o > 0; o >>= 1){ if (tid < o) red[tid] = fmaxf(red[tid], red[tid+o]); __syncthreads(); }
  float pmax = red[0]; __syncthreads();
  red[tid] = ss[tid]; __syncthreads();
  for (int o = 128; o > 0; o >>= 1){ if (tid < o) red[tid] = fmaxf(red[tid], red[tid+o]); __syncthreads(); }
  float smax = red[0]; __syncthreads();
  float v = fminf(ps[tid]/(pmax + 1e-6f), ss[tid]/(smax + 1e-6f));
  red[tid] = v; __syncthreads();
  for (int o = 128; o > 0; o >>= 1){ if (tid < o) red[tid] = fminf(red[tid], red[tid+o]); __syncthreads(); }
  if (tid == 0) an[b] = red[0];
}

// first-argmax over t of feat_mag rows, clamp to nc-1=9, pick scores
__global__ void select_kernel(const float* __restrict__ fm, const float* __restrict__ sc,
                              int* __restrict__ idxn, int* __restrict__ idxa,
                              float* __restrict__ sn, float* __restrict__ sa){
  int i = threadIdx.x;
  if (i >= 64) return;
  int half = i >> 5, ii = i & 31;
  const float* r = fm + (size_t)(half*32 + ii)*32;
  int arg = 0; float best = r[0];
  for (int t = 1; t < 32; ++t) if (r[t] > best){ best = r[t]; arg = t; }
  int id = arg < 9 ? arg : 9;
  if (half == 0){ idxn[ii] = id; sn[ii] = sc[(size_t)ii*32 + id]; }
  else          { idxa[ii] = id; sa[ii] = sc[(size_t)(32 + ii)*32 + id]; }
}

__global__ __launch_bounds__(256) void gather_kernel(const float* __restrict__ feat,
    const int* __restrict__ idxn, const int* __restrict__ idxa,
    float* __restrict__ tn, float* __restrict__ ta){
  long idx = (long)blockIdx.x*256 + threadIdx.x;   // 32*32*1024
  int f = (int)(idx & 1023);
  int t = (int)((idx >> 10) & 31);
  int i = (int)(idx >> 15);
  long rn =          (long)(i*10 + idxn[i])*32 + t;
  long ra = 10240 + ((long)(i*10 + idxa[i])*32 + t);
  tn[idx] = feat[rn*FF + f];
  ta[idx] = feat[ra*FF + f];
}

// ---------------------------------------------------------------------------
static inline void launch_gemm(hipStream_t st, const _Float16* A, const _Float16* Bt,
                               const float* bias, const float* resid,
                               float* C32, _Float16* C16,
                               int M, int N, int K, int ldc, int act, float scale){
  dim3 g((N + 127)/128, M/128), b(256);
  wmma_gemm<<<g, b, 0, st>>>(A, Bt, bias, resid, C32, C16, M, N, K, ldc, act, scale);
}

extern "C" void kernel_launch(void* const* d_in, const int* in_sizes, int n_in,
                              void* d_out, int out_size, void* d_ws, size_t ws_size,
                              hipStream_t stream)
{
  (void)in_sizes; (void)n_in; (void)out_size; (void)ws_size;

  // JAX pytree leaf order (dict keys sorted):
  // 0 inputs, 1 conv1_b, 2 conv1_w, 3 conv4_w, 4 conv5_w, 5 fc1_b, 6 fc1_w,
  // 7 fc2_b, 8 fc2_w, 9 fc3_b, 10 fc3_w, 11..22 m1{A,B,C,D,conv_b,conv_w,
  // in_b,in_w,out_b,out_w,ssm_b,ssm_w}, 23..34 m2{...}, 35 pmem, 36 recon_b,
  // 37 recon_w, 38 smem
  const float* inp     = (const float*)d_in[0];
  const float* conv1_b = (const float*)d_in[1];
  const float* conv1_w = (const float*)d_in[2];
  const float* conv4_w = (const float*)d_in[3];
  const float* conv5_w = (const float*)d_in[4];
  const float* fc1_b   = (const float*)d_in[5];
  const float* fc1_w   = (const float*)d_in[6];
  const float* fc2_b   = (const float*)d_in[7];
  const float* fc2_w   = (const float*)d_in[8];
  const float* fc3_b   = (const float*)d_in[9];
  const float* fc3_w   = (const float*)d_in[10];
  const float* m1p[12]; for (int i = 0; i < 12; ++i) m1p[i] = (const float*)d_in[11+i];
  const float* m2p[12]; for (int i = 0; i < 12; ++i) m2p[i] = (const float*)d_in[23+i];
  const float* pmem    = (const float*)d_in[35];
  const float* recon_b = (const float*)d_in[36];
  const float* recon_w = (const float*)d_in[37];
  const float* smem    = (const float*)d_in[38];
  // m-param indices: 0=A 1=B 2=C 3=D 4=conv_b 5=conv_w 6=in_b 7=in_w
  //                  8=out_b 9=out_w 10=ssm_b 11=ssm_w

  // ---- workspace layout ----
  char* wsb = (char*)d_ws;
  size_t off = 0;
  auto bump = [&](size_t bytes) -> void* {
    void* p = wsb + off;
    off += (bytes + 255) & ~(size_t)255;
    return p;
  };
  float*    XZ    = (float*)bump((size_t)NROW*2048*4);     // 160 MB (aliased by COL)
  _Float16* COL   = (_Float16*)XZ;                         // NROW*3072 f16 fits inside XZ
  float*    XA32  = (float*)bump((size_t)NROW*FF*4);
  _Float16* XA16  = (_Float16*)bump((size_t)NROW*FF*2);
  float*    XS    = (float*)bump((size_t)NROW*32*4);
  float*    CS    = (float*)bump((size_t)NROW*4);
  _Float16* YG    = (_Float16*)bump((size_t)NROW*FF*2);
  float*    B1O32 = (float*)bump((size_t)NROW*FF*4);
  _Float16* B1O16 = (_Float16*)bump((size_t)NROW*FF*2);
  float*    B2O32 = (float*)bump((size_t)NROW*FF*4);
  _Float16* B2O16 = (_Float16*)bump((size_t)NROW*FF*2);
  _Float16* CAT   = (_Float16*)bump((size_t)NROW*FF*2);
  float*    AG32  = (float*)bump((size_t)NROW*FF*4);
  _Float16* AG16  = (_Float16*)bump((size_t)NROW*FF*2);
  _Float16* S1    = (_Float16*)bump((size_t)NROW*512*2);
  _Float16* S2    = (_Float16*)bump((size_t)NROW*128*2);
  float*    S3    = (float*)bump((size_t)NROW*4);
  float*    MAG   = (float*)bump((size_t)NROW*4);
  int*      IDX   = (int*)bump(256);
  _Float16* X16   = (_Float16*)bump((size_t)NROW*FF*2);
  _Float16* W_IN1 = (_Float16*)bump((size_t)2048*1024*2);
  _Float16* W_OUT1= (_Float16*)bump((size_t)1024*1024*2);
  _Float16* W_SSM1= (_Float16*)bump((size_t)32*1024*2);
  _Float16* W_IN2 = (_Float16*)bump((size_t)2048*1024*2);
  _Float16* W_OUT2= (_Float16*)bump((size_t)1024*1024*2);
  _Float16* W_SSM2= (_Float16*)bump((size_t)32*1024*2);
  _Float16* W_C1  = (_Float16*)bump((size_t)512*3072*2);
  _Float16* W_C4  = (_Float16*)bump((size_t)512*1024*2);
  _Float16* W_C5  = (_Float16*)bump((size_t)1024*3072*2);
  _Float16* W_FC1 = (_Float16*)bump((size_t)512*1024*2);
  _Float16* W_FC2 = (_Float16*)bump((size_t)128*512*2);
  _Float16* W_REC = (_Float16*)bump((size_t)1024*1024*2);

  // ---- output section offsets (floats) ----
  float* out = (float*)d_out;
  const size_t O_SA = 0;
  const size_t O_SN = 32;
  const size_t O_TN = 64;
  const size_t O_TA = O_TN + (size_t)32*32*1024;
  const size_t O_SC = O_TA + (size_t)32*32*1024;
  const size_t O_FM = O_SC + 2048;
  const size_t O_AN = O_FM + 2048;
  const size_t O_RC = O_AN + 640;

  auto cvt = [&](const float* s, _Float16* d, long n){
    cvt16<<<dim3((unsigned)((n + 255)/256)), dim3(256), 0, stream>>>(s, d, n);
  };

  // ---- precision conversions ----
  cvt(inp, X16, (long)NROW*FF);
  cvt(m1p[7],  W_IN1,  (long)2048*1024);
  cvt(m1p[9],  W_OUT1, (long)1024*1024);
  cvt(m1p[11], W_SSM1, (long)32*1024);
  cvt(m2p[7],  W_IN2,  (long)2048*1024);
  cvt(m2p[9],  W_OUT2, (long)1024*1024);
  cvt(m2p[11], W_SSM2, (long)32*1024);
  cvt(conv4_w, W_C4,  (long)512*1024);
  cvt(fc1_w,   W_FC1, (long)512*1024);
  cvt(fc2_w,   W_FC2, (long)128*512);
  cvt(recon_w, W_REC, (long)1024*1024);
  cvt_convw<<<dim3((512*1024*3 + 255)/256), dim3(256), 0, stream>>>(conv1_w, W_C1, 512, 1024, 3);
  cvt_convw<<<dim3((1024*1024*3 + 255)/256), dim3(256), 0, stream>>>(conv5_w, W_C5, 1024, 1024, 3);

  const unsigned GEW = NROW*FF/256;   // 81920 element-wise blocks

  // ---- Mamba block 1 (residual = f32 inputs) ----
  launch_gemm(stream, X16, W_IN1, m1p[6], nullptr, XZ, nullptr, NROW, 2048, 1024, 2048, 0, 1.0f);
  mamba_mid<<<GEW, 256, 0, stream>>>(XZ, m1p[5], m1p[4], XA32, XA16);
  launch_gemm(stream, XA16, W_SSM1, m1p[10], nullptr, XS, nullptr, NROW, 32, 1024, 32, 0, 1.0f);
  scan_kernel<<<SEQ/8, 256, 0, stream>>>(XS, m1p[0], m1p[1], m1p[2], CS);
  ygate_kernel<<<GEW, 256, 0, stream>>>(CS, XA32, m1p[3], XZ, YG);
  launch_gemm(stream, YG, W_OUT1, m1p[8], inp, B1O32, B1O16, NROW, 1024, 1024, 1024, 0, 1.0f);

  // ---- Mamba block 2 (residual = block1 out) ----
  launch_gemm(stream, B1O16, W_IN2, m2p[6], nullptr, XZ, nullptr, NROW, 2048, 1024, 2048, 0, 1.0f);
  mamba_mid<<<GEW, 256, 0, stream>>>(XZ, m2p[5], m2p[4], XA32, XA16);
  launch_gemm(stream, XA16, W_SSM2, m2p[10], nullptr, XS, nullptr, NROW, 32, 1024, 32, 0, 1.0f);
  scan_kernel<<<SEQ/8, 256, 0, stream>>>(XS, m2p[0], m2p[1], m2p[2], CS);
  ygate_kernel<<<GEW, 256, 0, stream>>>(CS, XA32, m2p[3], XZ, YG);
  launch_gemm(stream, YG, W_OUT2, m2p[8], B1O32, B2O32, B2O16, NROW, 1024, 1024, 1024, 0, 1.0f);

  // ---- aggregate convs ----
  im2col3<<<NROW*384/256, 256, 0, stream>>>(B2O16, COL);
  // out1 = relu(conv1 + b) * BN_INV  -> cat cols [0,512)
  launch_gemm(stream, COL, W_C1, conv1_b, nullptr, nullptr, CAT, NROW, 512, 3072, 1024, 1, BN_INV);
  // out4 = relu(x @ conv4^T)         -> cat cols [512,1024)
  launch_gemm(stream, B2O16, W_C4, nullptr, nullptr, nullptr, CAT + 512, NROW, 512, 1024, 1024, 1, 1.0f);
  im2col3<<<NROW*384/256, 256, 0, stream>>>(CAT, COL);
  // out = relu(conv5) * BN_INV + residual(block2 out)
  launch_gemm(stream, COL, W_C5, nullptr, B2O32, AG32, AG16, NROW, 1024, 3072, 1024, 1, BN_INV);

  // ---- heads ----
  launch_gemm(stream, AG16, W_REC, recon_b, nullptr, out + O_RC, nullptr, NROW, 1024, 1024, 1024, 0, 1.0f);
  launch_gemm(stream, AG16, W_FC1, fc1_b, nullptr, nullptr, S1, NROW, 512, 1024, 512, 1, 1.0f);
  launch_gemm(stream, S1, W_FC2, fc2_b, nullptr, nullptr, S2, NROW, 128, 512, 128, 1, 1.0f);
  fc3_kernel<<<NROW/8, 256, 0, stream>>>(S2, fc3_w, fc3_b, S3);
  rownorm_kernel<<<NROW/8, 256, 0, stream>>>(AG32, MAG);
  scores_featmag<<<8, 256, 0, stream>>>(S3, MAG, out + O_SC, out + O_FM);
  anomaly_kernel<<<SEQ, 256, 0, stream>>>(AG32, pmem, smem, out + O_AN);
  select_kernel<<<1, 64, 0, stream>>>(out + O_FM, out + O_SC, IDX, IDX + 32,
                                      out + O_SN, out + O_SA);
  gather_kernel<<<32*32*1024/256, 256, 0, stream>>>(AG32, IDX, IDX + 32,
                                                    out + O_TN, out + O_TA);
}
